// SS2D_4879082848268
// MI455X (gfx1250) — matrix operand order repaired and checked by reading, not compiled, not run on hardware
//
#include <hip/hip_runtime.h>

// SS2D (VMamba selective-scan 2D) for gfx1250 / MI455X.
// B=2 H=W=64 DM=192 DI=384 DS=16 DR=12 L=4096, M_ROWS=B*L=8192.

#define LL 4096
#define DM_ 192
#define DI_ 384

typedef __attribute__((ext_vector_type(16))) __bf16 v16bf;
typedef __attribute__((ext_vector_type(8)))  float  v8f;
typedef int v4i __attribute__((vector_size(16)));
typedef __attribute__((address_space(1))) v4i* gv4i_p;   // global int4*
typedef __attribute__((address_space(3))) v4i* lv4i_p;   // LDS int4*

#if defined(__has_builtin)
#if __has_builtin(__builtin_amdgcn_global_load_async_to_lds_b128) && \
    __has_builtin(__builtin_amdgcn_s_wait_asynccnt)
#define HAVE_ASYNC_LDS 1
#endif
#if __has_builtin(__builtin_amdgcn_sched_group_barrier)
#define SCHED_GROUP(mask, size, id) __builtin_amdgcn_sched_group_barrier(mask, size, id)
#endif
#endif
#ifndef SCHED_GROUP
#define SCHED_GROUP(mask, size, id)
#endif

__device__ __forceinline__ unsigned short f2bu(float f) {
    union { float f; unsigned u; } a; a.f = f;
    unsigned u = a.u + 0x7fffu + ((a.u >> 16) & 1u);   // round-to-nearest-even
    return (unsigned short)(u >> 16);
}

#ifdef HAVE_ASYNC_LDS
// 16-byte async copy global -> LDS (ASYNCcnt-tracked, bypasses VGPRs).
__device__ __forceinline__ void async_cp16(const float* g, float* l) {
    __builtin_amdgcn_global_load_async_to_lds_b128(
        (gv4i_p)(g), (lv4i_p)(l), 0, 0);
}
#endif

// ---------------------------------------------------------------------------
// One-shot prep: convert x + all weights to bf16; zero-pad x_proj_w rows
// (44 -> 48) and dt_w K (12 -> 32) so GEMMs need no bounds predicates.
// ---------------------------------------------------------------------------
__global__ __launch_bounds__(256)
void prep_bf16(const float* __restrict__ x, const float* __restrict__ in_w,
               const float* __restrict__ xproj_w, const float* __restrict__ dt_w,
               const float* __restrict__ out_w,
               unsigned short* __restrict__ xb, unsigned short* __restrict__ in_wb,
               unsigned short* __restrict__ xprojb, unsigned short* __restrict__ dtb,
               unsigned short* __restrict__ outb)
{
    int i = blockIdx.x * 256 + threadIdx.x;
    const int n_x  = 8192 * 192;
    const int n_in = 768 * 192;
    const int n_xp = 48 * 384;
    const int n_dt = 384 * 32;
    const int n_out = 192 * 384;
    if (i < n_x) { xb[i] = f2bu(x[i]); return; }
    i -= n_x;
    if (i < n_in) { in_wb[i] = f2bu(in_w[i]); return; }
    i -= n_in;
    if (i < n_xp) {
        int n = i / 384, k = i - n * 384;
        xprojb[i] = (n < 44) ? f2bu(xproj_w[n * 384 + k]) : (unsigned short)0;
        return;
    }
    i -= n_xp;
    if (i < n_dt) {
        int n = i / 32, k = i - n * 32;
        dtb[i] = (k < 12) ? f2bu(dt_w[n * 12 + k]) : (unsigned short)0;
        return;
    }
    i -= n_dt;
    if (i < n_out) outb[i] = f2bu(out_w[i]);
}

// ---------------------------------------------------------------------------
// bf16 WMMA GEMM: C(MxN) = A(MxK) * W(NxK)^T, f32 accumulate. All shapes are
// compile-time; fragments are 2x global_load_b128; K-loop is double-buffered
// and sched_group_barrier pins the pipeline: step k+1's loads issue before
// step k's WMMAs, so WMMA waits are partial, not full drains.
// Block = 256 thr = 8 waves; wave tile = 16(M) x 16*NW(N); block = 128 x 16*NW.
// grid = (N/(16*NW), M/128).
// MODE 0: f32 store. MODE 1: softplus(acc+bias[n]) scattered to alt[b,d=n,l].
// MODE 2: f32 store + bf16 copy to altb (same layout).
// ---------------------------------------------------------------------------
template <int LDA, int LDW, int LDC, int K, int NW, int MODE>
__global__ __launch_bounds__(256)
void gemm_wmma(const unsigned short* __restrict__ A,
               const unsigned short* __restrict__ Wt,
               float* __restrict__ C,
               const float* __restrict__ bias,
               float* __restrict__ alt,
               unsigned short* __restrict__ altb)
{
    const int lane  = threadIdx.x & 31;
    const int wv    = threadIdx.x >> 5;
    const int tileM = blockIdx.y * 128 + wv * 16;
    const int tileN = blockIdx.x * (16 * NW);
    const int mh    = lane & 15;
    const bool hi   = lane >= 16;
    const int kbA   = hi ? 8 : 0;     // A 16x32: lanes hold K {kb..kb+7, kb+16..kb+23}
    const int kbB   = hi ? 8 : 0;     // B 32x16: lanes hold K {kb..kb+7, kb+16..kb+23}

    v8f acc[NW];
#pragma unroll
    for (int t = 0; t < NW; ++t) acc[t] = (v8f){0.f,0.f,0.f,0.f,0.f,0.f,0.f,0.f};

    union Frag { v16bf v; uint4 q[2]; };
    Frag a0, a1, b0[NW], b1[NW];

    const unsigned short* arow  = A  + (size_t)(tileM + mh) * LDA + kbA;
    const unsigned short* wbase = Wt + (size_t)(tileN + mh) * LDW + kbB;

    a0.q[0] = *(const uint4*)(arow);
    a0.q[1] = *(const uint4*)(arow + 16);
#pragma unroll
    for (int t = 0; t < NW; ++t) {
        b0[t].q[0] = *(const uint4*)(wbase + t * 16 * LDW);
        b0[t].q[1] = *(const uint4*)(wbase + t * 16 * LDW + 16);
    }

#pragma unroll
    for (int k0 = 0; k0 < K; k0 += 32) {
        if (k0 + 32 < K) {                       // prefetch next K-step
            a1.q[0] = *(const uint4*)(arow + k0 + 32);
            a1.q[1] = *(const uint4*)(arow + k0 + 48);
#pragma unroll
            for (int t = 0; t < NW; ++t) {
                b1[t].q[0] = *(const uint4*)(wbase + t * 16 * LDW + k0 + 32);
                b1[t].q[1] = *(const uint4*)(wbase + t * 16 * LDW + k0 + 48);
            }
        }
#pragma unroll
        for (int t = 0; t < NW; ++t)
            acc[t] = __builtin_amdgcn_wmma_f32_16x16x32_bf16(
                         false, a0.v, false, b0[t].v, (short)0, acc[t], false, false);
        if (k0 + 32 < K) {
            // Pin the pipeline: next step's VMEM reads, then this step's WMMAs.
            SCHED_GROUP(0x020, 2 + 2 * NW, 0);   // VMEM read group
            SCHED_GROUP(0x008, NW, 0);           // WMMA group
            a0 = a1;
#pragma unroll
            for (int t = 0; t < NW; ++t) b0[t] = b1[t];
        }
    }

    // C layout: VGPR r -> M = r (lanes 0-15) / r+8 (lanes 16-31), N = lane&15.
    const int mbase = tileM + (hi ? 8 : 0);
#pragma unroll
    for (int t = 0; t < NW; ++t) {
        const int ncol = tileN + t * 16 + mh;
#pragma unroll
        for (int r = 0; r < 8; ++r) {
            const int m = mbase + r;
            const float v = acc[t][r];
            if (MODE == 0) {
                C[(size_t)m * LDC + ncol] = v;
            } else if (MODE == 2) {
                C[(size_t)m * LDC + ncol] = v;
                altb[(size_t)m * LDC + ncol] = f2bu(v);
            } else {
                float xv = v + bias[ncol];
                float sp = (xv > 20.f) ? xv : log1pf(__expf(xv));
                int b = m >> 12, l = m & (LL - 1);
                alt[((size_t)b * DI_ + ncol) * LL + l] = sp;   // delta (B,DI,L)
            }
        }
    }
}

// ---------------------------------------------------------------------------
// Depthwise 3x3 conv (SAME) + bias + SiLU + zigzag; writes f32 u (B,DI,L) and
// bf16 transposed copy (B*L, DI) for the x_proj WMMA.
// ---------------------------------------------------------------------------
__global__ __launch_bounds__(128)
void conv_silu_zig(const float* __restrict__ xz,   // (B*L, 768), xc = [:384]
                   const float* __restrict__ cw,   // (DI,1,3,3)
                   const float* __restrict__ cb,
                   float* __restrict__ xs, unsigned short* __restrict__ xsT)
{
    int pix = blockIdx.x;
    int b = pix >> 12;
    int hw = pix & 4095;
    int h = hw >> 6, w = hw & 63;
    int lseq = (h & 1) ? (h * 64 + (63 - w)) : hw;       // zigzag
    size_t rowT = ((size_t)b * LL + lseq) * DI_;
#pragma unroll
    for (int j = 0; j < 3; ++j) {
        int d = threadIdx.x + 128 * j;
        float accv = cb[d];
#pragma unroll
        for (int kh = 0; kh < 3; ++kh) {
            int h2 = h + kh - 1;
            if (h2 < 0 || h2 >= 64) continue;
#pragma unroll
            for (int kw = 0; kw < 3; ++kw) {
                int w2 = w + kw - 1;
                if (w2 < 0 || w2 >= 64) continue;
                float v = xz[((size_t)b * LL + h2 * 64 + w2) * 768 + d];
                accv += v * cw[d * 9 + kh * 3 + kw];
            }
        }
        float s = accv / (1.f + __expf(-accv));          // SiLU
        xs[((size_t)b * DI_ + d) * LL + lseq] = s;
        xsT[rowT + d] = f2bu(s);
    }
}

// ---------------------------------------------------------------------------
// Selective scan. Block = 256 thr = 8 waves; wave = 2 channels x 16 states.
// B/C/delta/u staged into LDS in 32-step chunks via async global->LDS b128
// copies (ASYNCcnt) when available, else vectorized float4 staging. State
// reduction via __shfl_xor. Writes y un-zigzagged into (B, spatial, DI) + u*D.
// ---------------------------------------------------------------------------
__global__ __launch_bounds__(256)
void scan_kernel(const float* __restrict__ x_dbl,  // (B*L, 48): dt|B|C
                 const float* __restrict__ delta,  // (B, DI, L)
                 const float* __restrict__ xs,     // (B, DI, L)
                 const float* __restrict__ A_logs, // (DI, 16)
                 const float* __restrict__ Dvec,   // (DI)
                 float* __restrict__ yT)           // (B*L_spatial, DI)
{
    const int CH = 32;
    __shared__ __align__(16) float sB[CH * 16];
    __shared__ __align__(16) float sC[CH * 16];
    __shared__ __align__(16) float sD[16 * CH];
    __shared__ __align__(16) float sU[16 * CH];

    int blk = blockIdx.x;
    int b = blk / 24;
    int dbase = (blk % 24) * 16;
    int tid = threadIdx.x;
    int lane = tid & 31;
    int wv = tid >> 5;
    int s = lane & 15;
    int dloc = 2 * wv + (lane >> 4);
    int d = dbase + dloc;

    float Av = -__expf(A_logs[d * 16 + s]);
    float Dd = Dvec[d];
    float h = 0.f;

    const float* dl_base = delta + ((size_t)b * DI_ + dbase) * LL;
    const float* u_base  = xs    + ((size_t)b * DI_ + dbase) * LL;

    // Per-chunk staging assignment (each thread moves 2x16 bytes):
    //  tid 0..127   -> sB quad + sD quad ; tid 128..255 -> sC quad + sU quad.
    const int t7   = tid & 127;
    const int bi   = t7 >> 2;             // row 0..31 of B/C chunk
    const int bc   = (t7 & 3) * 4;        // col 0,4,8,12
    const int ddst = t7 >> 3;             // channel 0..15 of D/U chunk
    const int dcol = (t7 & 7) * 4;        // l-offset 0..28
    const bool hiT = tid >= 128;

    for (int l0 = 0; l0 < LL; l0 += CH) {
        {
            const float* gBC = x_dbl + ((size_t)b * LL + l0 + bi) * 48 + (hiT ? 28 : 12) + bc;
            float*       lBC = (hiT ? sC : sB) + bi * 16 + bc;
            const float* gDU = (hiT ? u_base : dl_base) + (size_t)ddst * LL + l0 + dcol;
            float*       lDU = (hiT ? sU : sD) + ddst * CH + dcol;
#ifdef HAVE_ASYNC_LDS
            async_cp16(gBC, lBC);
            async_cp16(gDU, lDU);
            __builtin_amdgcn_s_wait_asynccnt(0);
#else
            *(float4*)lBC = *(const float4*)gBC;
            *(float4*)lDU = *(const float4*)gDU;
#endif
        }
        __syncthreads();
        for (int i = 0; i < CH; ++i) {
            float dl = sD[dloc * CH + i];
            float u  = sU[dloc * CH + i];
            float dA = __expf(dl * Av);
            h = dA * h + (dl * u) * sB[i * 16 + s];
            float p = h * sC[i * 16 + s];
            p += __shfl_xor(p, 1, 32);
            p += __shfl_xor(p, 2, 32);
            p += __shfl_xor(p, 4, 32);
            p += __shfl_xor(p, 8, 32);
            if (s == 0) {
                int l = l0 + i;
                int row = l >> 6, col = l & 63;
                int colsp = (row & 1) ? (63 - col) : col;      // un-zigzag
                int spat = (row << 6) | colsp;
                yT[((size_t)b * LL + spat) * DI_ + d] = p + u * Dd;
            }
        }
        __syncthreads();
    }
}

// ---------------------------------------------------------------------------
// LayerNorm over DI + SiLU gate; emits bf16 A-matrix rows for out_proj.
// ---------------------------------------------------------------------------
__global__ __launch_bounds__(128)
void ln_gate(const float* __restrict__ yT, const float* __restrict__ xz,
             const float* __restrict__ lnw, const float* __restrict__ lnb,
             unsigned short* __restrict__ g)
{
    int row = blockIdx.x;                 // b*L + spatial
    int tid = threadIdx.x;
    __shared__ float ssum[4], ssq[4];
    float v[3]; float s = 0.f, q = 0.f;
#pragma unroll
    for (int j = 0; j < 3; ++j) {
        v[j] = yT[(size_t)row * DI_ + tid + 128 * j];
        s += v[j]; q += v[j] * v[j];
    }
#pragma unroll
    for (int m = 16; m >= 1; m >>= 1) {
        s += __shfl_xor(s, m, 32);
        q += __shfl_xor(q, m, 32);
    }
    if ((tid & 31) == 0) { ssum[tid >> 5] = s; ssq[tid >> 5] = q; }
    __syncthreads();
    s = ssum[0] + ssum[1] + ssum[2] + ssum[3];
    q = ssq[0] + ssq[1] + ssq[2] + ssq[3];
    float mu  = s * (1.f / DI_);
    float var = q * (1.f / DI_) - mu * mu;
    float rstd = rsqrtf(var + 1e-5f);
#pragma unroll
    for (int j = 0; j < 3; ++j) {
        int c = tid + 128 * j;
        float z  = xz[(size_t)row * 768 + DI_ + c];
        float gn = (v[j] - mu) * rstd * lnw[c] + lnb[c];
        float o  = gn * (z / (1.f + __expf(-z)));
        g[(size_t)row * DI_ + c] = f2bu(o);
    }
}

// ---------------------------------------------------------------------------
extern "C" void kernel_launch(void* const* d_in, const int* in_sizes, int n_in,
                              void* d_out, int out_size, void* d_ws, size_t ws_size,
                              hipStream_t stream)
{
    (void)in_sizes; (void)n_in; (void)out_size; (void)ws_size;
    const float* x       = (const float*)d_in[0];
    const float* in_w    = (const float*)d_in[1];
    const float* conv_w  = (const float*)d_in[2];
    const float* conv_b  = (const float*)d_in[3];
    const float* xproj_w = (const float*)d_in[4];
    const float* dt_w    = (const float*)d_in[5];
    const float* dt_b    = (const float*)d_in[6];
    const float* A_logs  = (const float*)d_in[7];
    const float* Dvec    = (const float*)d_in[8];
    const float* ln_w    = (const float*)d_in[9];
    const float* ln_b    = (const float*)d_in[10];
    const float* out_w   = (const float*)d_in[11];

    char* ws = (char*)d_ws;
    float*          xz     = (float*)(ws);                     // (8192,768) f32
    float*          xs     = (float*)(ws + 25165824);          // (B,DI,L)  f32
    unsigned short* xsT    = (unsigned short*)(ws + 37748736); // (8192,384) bf16
    float*          x_dbl  = (float*)(ws + 44040192);          // (8192,48) f32
    unsigned short* x_dblb = (unsigned short*)(ws + 45613056); // (8192,48) bf16
    float*          delta  = (float*)(ws + 46399488);          // (B,DI,L)  f32
    float*          yT     = (float*)(ws + 58982400);          // (8192,384) f32
    unsigned short* xb     = (unsigned short*)(ws + 71565312); // (8192,192) bf16
    unsigned short* in_wb  = (unsigned short*)(ws + 74711040); // (768,192) bf16
    unsigned short* xprojb = (unsigned short*)(ws + 75005952); // (48,384)  bf16
    unsigned short* dtb    = (unsigned short*)(ws + 75042816); // (384,32)  bf16
    unsigned short* outb   = (unsigned short*)(ws + 75067392); // (192,384) bf16
    unsigned short* g      = xsT;                              // reuse after x_proj

    dim3 blk(256);
    // 0) bf16 conversion + zero-padding of operands
    prep_bf16<<<dim3((1824768 + 255) / 256), blk, 0, stream>>>(
        x, in_w, xproj_w, dt_w, out_w, xb, in_wb, xprojb, dtb, outb);
    // 1) in_proj: (8192x192) x (768x192)^T -> xz
    gemm_wmma<192, 192, 768, 192, 4, 0><<<dim3(768 / 64, 64), blk, 0, stream>>>(
        xb, in_wb, xz, nullptr, nullptr, nullptr);
    // 2) depthwise conv + SiLU + zigzag
    conv_silu_zig<<<dim3(8192), dim3(128), 0, stream>>>(xz, conv_w, conv_b, xs, xsT);
    // 3) x_proj: (8192x384) x (48x384)^T -> x_dbl (f32 + bf16 copy)
    gemm_wmma<384, 384, 48, 384, 3, 2><<<dim3(1, 64), blk, 0, stream>>>(
        xsT, xprojb, x_dbl, nullptr, nullptr, x_dblb);
    // 4) dt proj (K padded 12->32) + bias + softplus -> delta (B,DI,L)
    gemm_wmma<48, 32, 1, 32, 4, 1><<<dim3(384 / 64, 64), blk, 0, stream>>>(
        x_dblb, dtb, nullptr, dt_b, delta, nullptr);
    // 5) selective scan -> yT (un-zigzagged, (B*L, DI))
    scan_kernel<<<dim3(48), dim3(256), 0, stream>>>(x_dbl, delta, xs, A_logs, Dvec, yT);
    // 6) LayerNorm + SiLU gate -> g bf16
    ln_gate<<<dim3(8192), dim3(128), 0, stream>>>(yT, xz, ln_w, ln_b, g);
    // 7) out_proj: (8192x384) x (192x384)^T -> d_out
    gemm_wmma<384, 384, 192, 384, 4, 0><<<dim3(192 / 16 / 4, 64), blk, 0, stream>>>(
        g, outb, (float*)d_out, nullptr, nullptr, nullptr);
}
